// resconv_basic_43516608643443
// MI455X (gfx1250) — compile-verified, matrix operand order checked by reading, not run on hardware
//
#include <hip/hip_runtime.h>
#include <hip/hip_bf16.h>
#include <stdint.h>

// ---------------------------------------------------------------------------
// MI455X (gfx1250) fused SelectionConv residual block.
//   conv1: CSR-gather(x) -> [16x1152] LDS tile -> bf16 WMMA GEMM -> BN+ELU -> h (bf16)
//   conv2+sc: CSR-gather(h) + sel0-gather(x) -> [16x1280] tile -> 2 WMMA GEMMs
//             -> elu(bn2(.)) + (sc+b3) -> elu(bn3(.)) -> out (f32)
// CSR built on device (histogram / scan / scatter). Weights pre-swizzled into
// per-lane WMMA B-fragment order so the GEMM inner loop is pure vector loads
// + v_wmma_f32_16x16x32_bf16 (f32 accumulate).
// ---------------------------------------------------------------------------

typedef __attribute__((ext_vector_type(16))) __bf16        v16bf;
typedef __attribute__((ext_vector_type(8)))  float         v8f;
typedef __attribute__((ext_vector_type(4)))  unsigned int  u32x4;

#define NN     65536
#define CH     128
#define NDIRS  9
#define NE     589824
#define KAGG1  (NDIRS * CH)   /* 1152 */
#define KT1    (KAGG1 / 32)   /* 36   */
#define KAGG2  (KAGG1 + CH)   /* 1280 */
#define KT3    (CH / 32)      /* 4    */
#define NTILES (CH / 16)      /* 8    */

__device__ __forceinline__ unsigned short f2bf(float f) {
  unsigned int u = __float_as_uint(f);
  u += 0x7FFFu + ((u >> 16) & 1u);          // round-to-nearest-even
  return (unsigned short)(u >> 16);
}
__device__ __forceinline__ float elu1(float y) {
  return y > 0.f ? y : (__expf(y) - 1.f);
}

// ------------------------------- CSR build ---------------------------------

__global__ void zero_i32(int* __restrict__ p, int n) {
  int i = blockIdx.x * 256 + threadIdx.x;
  if (i < n) p[i] = 0;
}

__global__ void hist_kernel(const int* __restrict__ dst, int* __restrict__ counts) {
  int e = blockIdx.x * 256 + threadIdx.x;
  if (e < NE) atomicAdd(&counts[dst[e]], 1);
}

__global__ __launch_bounds__(1024) void scan_kernel(const int* __restrict__ counts,
                                                    int* __restrict__ row_ptr,
                                                    int* __restrict__ cursor) {
  __shared__ int part[1024];
  const int t = threadIdx.x;
  const int per = NN / 1024;  // 64
  const int base = t * per;
  int s = 0;
  for (int i = 0; i < per; ++i) s += counts[base + i];
  part[t] = s;
  __syncthreads();
  for (int off = 1; off < 1024; off <<= 1) {
    int v = (t >= off) ? part[t - off] : 0;
    __syncthreads();
    part[t] += v;
    __syncthreads();
  }
  int run = part[t] - s;  // exclusive prefix for this thread's chunk
  for (int i = 0; i < per; ++i) {
    row_ptr[base + i] = run;
    cursor[base + i]  = run;
    run += counts[base + i];
  }
  if (t == 1023) row_ptr[NN] = run;  // == NE
}

__global__ void scatter_kernel(const int* __restrict__ src, const int* __restrict__ dst,
                               const int* __restrict__ sel, int* __restrict__ cursor,
                               unsigned int* __restrict__ csr) {
  int e = blockIdx.x * 256 + threadIdx.x;
  if (e >= NE) return;
  int d = dst[e];
  int pos = atomicAdd(&cursor[d], 1);
  csr[pos] = (((unsigned)sel[e]) << 16) | ((unsigned)src[e] & 0xFFFFu);  // N==2^16
}

// ------------------- weight pre-swizzle into B-fragment order --------------
// Fragment for (kt, nt, lane): column n = nt*16 + (lane&15);
// element e -> k = kt*32 + (lane>>4)*8 + (e<8 ? e : e+8)

__global__ void wprep_kernel(const float* __restrict__ W1, const float* __restrict__ W2,
                             const float* __restrict__ W3,
                             unsigned short* __restrict__ Wf1,
                             unsigned short* __restrict__ Wf2,
                             unsigned short* __restrict__ Wf3) {
  int tid  = blockIdx.x * 256 + threadIdx.x;
  int lane = tid & 31;
  int nt   = (tid >> 5) & 7;
  int t    = tid >> 8;  // global k-tile id across all three weights
  const float* Wsrc;
  unsigned short* Wdst;
  int kt;
  if (t < KT1)            { Wsrc = W1; Wdst = Wf1; kt = t; }
  else if (t < 2 * KT1)   { Wsrc = W2; Wdst = Wf2; kt = t - KT1; }
  else                    { Wsrc = W3; Wdst = Wf3; kt = t - 2 * KT1; }
  const int n    = nt * 16 + (lane & 15);
  const int koff = (lane >> 4) * 8;
  unsigned short frag[16];
#pragma unroll
  for (int e = 0; e < 16; ++e) {
    int k = kt * 32 + koff + (e < 8 ? e : e + 8);
    frag[e] = f2bf(Wsrc[(size_t)k * CH + n]);
  }
  unsigned short* o = Wdst + ((size_t)(kt * NTILES + nt) * 32 + lane) * 16;
#pragma unroll
  for (int e = 0; e < 16; ++e) o[e] = frag[e];
}

// -------------------------- fused conv kernels -----------------------------

union Frag { v16bf v; u32x4 u[2]; };

__device__ __forceinline__ void load_bfrag(Frag& b, const unsigned short* Wf,
                                           int kt, int nt, int lane) {
  const u32x4* wp = (const u32x4*)(Wf + ((size_t)(kt * NTILES + nt) * 32 + lane) * 16);
  b.u[0] = wp[0];
  b.u[1] = wp[1];
}

// conv1: x -> h (bf16). Tile: 16 nodes, K = 1152.
__global__ __launch_bounds__(128) void conv1_kernel(
    const float* __restrict__ x,
    const int* __restrict__ row_ptr, const unsigned int* __restrict__ csr,
    const unsigned short* __restrict__ Wf,
    const float* __restrict__ b, const float* __restrict__ g,
    const float* __restrict__ be, const float* __restrict__ rm,
    const float* __restrict__ rv,
    unsigned short* __restrict__ hout) {
  constexpr int KAGG = KAGG1;      // 1152
  constexpr int KP   = KAGG + 8;   // padded bf16 row stride (4-bank skew)
  extern __shared__ char smem[];
  float* Af = (float*)smem;                                       // [16][KAGG] f32
  unsigned short* Ab = (unsigned short*)(smem + 16 * KAGG * 4);   // [16][KP] bf16
  const int tid = threadIdx.x;

  for (int i = tid; i < 16 * KAGG; i += 128) Af[i] = 0.f;
  __syncthreads();

  {  // gather: thread owns (node nl, 16-channel slice c0) -> no collisions
    const int nl = tid >> 3, c0 = (tid & 7) << 4;
    const int node = blockIdx.x * 16 + nl;
    const int e1 = row_ptr[node + 1];
    float* arow = Af + nl * KAGG;
    for (int e = row_ptr[node]; e < e1; ++e) {
      unsigned p = csr[e];
      int src = p & 0xFFFF, sel = p >> 16;
      const float4* xp = (const float4*)(x + (size_t)src * CH + c0);
      float* a = arow + sel * CH + c0;
#pragma unroll
      for (int q = 0; q < 4; ++q) {
        float4 v = xp[q];
        a[4 * q + 0] += v.x; a[4 * q + 1] += v.y;
        a[4 * q + 2] += v.z; a[4 * q + 3] += v.w;
      }
    }
  }
  __syncthreads();

  for (int i = tid; i < 16 * KAGG; i += 128) {  // f32 -> bf16 with padded stride
    int r = i / KAGG, k = i - r * KAGG;
    Ab[r * KP + k] = f2bf(Af[i]);
  }
  __syncthreads();

  const int lane = tid & 31, wave = tid >> 5;
  const int m = lane & 15, koff = (lane >> 4) * 8;
  v8f acc0 = {0, 0, 0, 0, 0, 0, 0, 0};
  v8f acc1 = {0, 0, 0, 0, 0, 0, 0, 0};
  for (int kt = 0; kt < KT1; ++kt) {
    Frag a, b0, b1;
    const unsigned short* ap = Ab + m * KP + kt * 32 + koff;
    a.u[0] = *(const u32x4*)ap;
    a.u[1] = *(const u32x4*)(ap + 16);
    load_bfrag(b0, Wf, kt, wave * 2 + 0, lane);
    load_bfrag(b1, Wf, kt, wave * 2 + 1, lane);
    acc0 = __builtin_amdgcn_wmma_f32_16x16x32_bf16(false, a.v, false, b0.v,
                                                   (short)0, acc0, false, false);
    acc1 = __builtin_amdgcn_wmma_f32_16x16x32_bf16(false, a.v, false, b1.v,
                                                   (short)0, acc1, false, false);
  }

  const int mo0 = (lane >> 4) * 8;
#pragma unroll
  for (int j = 0; j < 2; ++j) {
    const int n = (wave * 2 + j) * 16 + (lane & 15);
    const float sc = g[n] * rsqrtf(rv[n] + 1e-5f);
    const float bb = b[n], rmn = rm[n], ben = be[n];
    v8f acc = j ? acc1 : acc0;
#pragma unroll
    for (int r = 0; r < 8; ++r) {
      float y = acc[r] + bb;
      y = (y - rmn) * sc + ben;
      hout[(size_t)(blockIdx.x * 16 + mo0 + r) * CH + n] = f2bf(elu1(y));
    }
  }
}

// conv2 + 1x1 shortcut + BN3 + ELU -> out (f32). Tile K = 1152 (h) + 128 (x, sel==0).
__global__ __launch_bounds__(128) void conv2_kernel(
    const unsigned short* __restrict__ h, const float* __restrict__ x,
    const int* __restrict__ row_ptr, const unsigned int* __restrict__ csr,
    const unsigned short* __restrict__ Wf2, const unsigned short* __restrict__ Wf3,
    const float* __restrict__ b2, const float* __restrict__ g2,
    const float* __restrict__ be2, const float* __restrict__ rm2,
    const float* __restrict__ rv2,
    const float* __restrict__ b3, const float* __restrict__ g3,
    const float* __restrict__ be3, const float* __restrict__ rm3,
    const float* __restrict__ rv3,
    float* __restrict__ out) {
  constexpr int KAGG = KAGG2;      // 1280
  constexpr int KP   = KAGG + 8;   // 1288
  extern __shared__ char smem[];
  float* Af = (float*)smem;
  unsigned short* Ab = (unsigned short*)(smem + 16 * KAGG * 4);
  const int tid = threadIdx.x;

  for (int i = tid; i < 16 * KAGG; i += 128) Af[i] = 0.f;
  __syncthreads();

  {
    const int nl = tid >> 3, c0 = (tid & 7) << 4;
    const int node = blockIdx.x * 16 + nl;
    const int e1 = row_ptr[node + 1];
    float* arow = Af + nl * KAGG;
    for (int e = row_ptr[node]; e < e1; ++e) {
      unsigned p = csr[e];
      int src = p & 0xFFFF, sel = p >> 16;
      const u32x4* hp = (const u32x4*)(h + (size_t)src * CH + c0);
      u32x4 v0 = hp[0], v1 = hp[1];
      float* a = arow + sel * CH + c0;
#pragma unroll
      for (int w = 0; w < 4; ++w) {
        unsigned u = v0[w];
        a[2 * w + 0] += __uint_as_float(u << 16);
        a[2 * w + 1] += __uint_as_float(u & 0xFFFF0000u);
      }
#pragma unroll
      for (int w = 0; w < 4; ++w) {
        unsigned u = v1[w];
        a[8 + 2 * w + 0] += __uint_as_float(u << 16);
        a[8 + 2 * w + 1] += __uint_as_float(u & 0xFFFF0000u);
      }
      if (sel == 0) {  // 1x1 shortcut aggregates x over selection-0 edges
        const float4* xp = (const float4*)(x + (size_t)src * CH + c0);
        float* asc = arow + KAGG1 + c0;
#pragma unroll
        for (int q = 0; q < 4; ++q) {
          float4 v = xp[q];
          asc[4 * q + 0] += v.x; asc[4 * q + 1] += v.y;
          asc[4 * q + 2] += v.z; asc[4 * q + 3] += v.w;
        }
      }
    }
  }
  __syncthreads();

  for (int i = tid; i < 16 * KAGG; i += 128) {
    int r = i / KAGG, k = i - r * KAGG;
    Ab[r * KP + k] = f2bf(Af[i]);
  }
  __syncthreads();

  const int lane = tid & 31, wave = tid >> 5;
  const int m = lane & 15, koff = (lane >> 4) * 8;
  v8f aM0 = {0, 0, 0, 0, 0, 0, 0, 0}, aM1 = {0, 0, 0, 0, 0, 0, 0, 0};
  v8f aS0 = {0, 0, 0, 0, 0, 0, 0, 0}, aS1 = {0, 0, 0, 0, 0, 0, 0, 0};
  for (int kt = 0; kt < KT1; ++kt) {  // main conv2 GEMM over h-aggregate
    Frag a, b0, b1;
    const unsigned short* ap = Ab + m * KP + kt * 32 + koff;
    a.u[0] = *(const u32x4*)ap;
    a.u[1] = *(const u32x4*)(ap + 16);
    load_bfrag(b0, Wf2, kt, wave * 2 + 0, lane);
    load_bfrag(b1, Wf2, kt, wave * 2 + 1, lane);
    aM0 = __builtin_amdgcn_wmma_f32_16x16x32_bf16(false, a.v, false, b0.v,
                                                  (short)0, aM0, false, false);
    aM1 = __builtin_amdgcn_wmma_f32_16x16x32_bf16(false, a.v, false, b1.v,
                                                  (short)0, aM1, false, false);
  }
  for (int kt = 0; kt < KT3; ++kt) {  // shortcut GEMM over x-aggregate
    Frag a, b0, b1;
    const unsigned short* ap = Ab + m * KP + KAGG1 + kt * 32 + koff;
    a.u[0] = *(const u32x4*)ap;
    a.u[1] = *(const u32x4*)(ap + 16);
    load_bfrag(b0, Wf3, kt, wave * 2 + 0, lane);
    load_bfrag(b1, Wf3, kt, wave * 2 + 1, lane);
    aS0 = __builtin_amdgcn_wmma_f32_16x16x32_bf16(false, a.v, false, b0.v,
                                                  (short)0, aS0, false, false);
    aS1 = __builtin_amdgcn_wmma_f32_16x16x32_bf16(false, a.v, false, b1.v,
                                                  (short)0, aS1, false, false);
  }

  const int mo0 = (lane >> 4) * 8;
#pragma unroll
  for (int j = 0; j < 2; ++j) {
    const int n = (wave * 2 + j) * 16 + (lane & 15);
    const float s2 = g2[n] * rsqrtf(rv2[n] + 1e-5f);
    const float s3 = g3[n] * rsqrtf(rv3[n] + 1e-5f);
    const float b2n = b2[n], rm2n = rm2[n], be2n = be2[n];
    const float b3n = b3[n], rm3n = rm3[n], be3n = be3[n];
    v8f accM = j ? aM1 : aM0;
    v8f accS = j ? aS1 : aS0;
#pragma unroll
    for (int r = 0; r < 8; ++r) {
      float y2 = elu1((accM[r] + b2n - rm2n) * s2 + be2n);  // elu(bn2(conv2+b2))
      float sc = accS[r] + b3n;                              // conv3 shortcut + b3
      float yo = ((y2 + sc) - rm3n) * s3 + be3n;             // bn3
      out[(size_t)(blockIdx.x * 16 + mo0 + r) * CH + n] = elu1(yo);
    }
  }
}

// ------------------------------- launcher ----------------------------------

extern "C" void kernel_launch(void* const* d_in, const int* in_sizes, int n_in,
                              void* d_out, int out_size, void* d_ws, size_t ws_size,
                              hipStream_t stream) {
  (void)in_sizes; (void)n_in; (void)out_size; (void)ws_size;
  const float* x   = (const float*)d_in[0];
  const int* eidx  = (const int*)d_in[1];
  const int* sel   = (const int*)d_in[2];
  const float* W1  = (const float*)d_in[3];
  const float* b1  = (const float*)d_in[4];
  const float* g1  = (const float*)d_in[5];
  const float* be1 = (const float*)d_in[6];
  const float* rm1 = (const float*)d_in[7];
  const float* rv1 = (const float*)d_in[8];
  const float* W2  = (const float*)d_in[9];
  const float* b2  = (const float*)d_in[10];
  const float* g2  = (const float*)d_in[11];
  const float* be2 = (const float*)d_in[12];
  const float* rm2 = (const float*)d_in[13];
  const float* rv2 = (const float*)d_in[14];
  const float* W3  = (const float*)d_in[15];
  const float* b3  = (const float*)d_in[16];
  const float* g3  = (const float*)d_in[17];
  const float* be3 = (const float*)d_in[18];
  const float* rm3 = (const float*)d_in[19];
  const float* rv3 = (const float*)d_in[20];

  const int* srcv = eidx;        // edge_index[0]
  const int* dstv = eidx + NE;   // edge_index[1]

  // workspace layout (bytes, 256-aligned), total ~20.5 MB
  char* ws = (char*)d_ws;
  int*            counts  = (int*)(ws + 0);               // 256 KB
  int*            row_ptr = (int*)(ws + 262144);          // 256 KB (+4)
  int*            cursor  = (int*)(ws + 524544);          // 256 KB
  unsigned int*   csr     = (unsigned int*)(ws + 786688); // 2.25 MB
  unsigned short* Wf1     = (unsigned short*)(ws + 3145984);
  unsigned short* Wf2     = (unsigned short*)(ws + 3440896);
  unsigned short* Wf3     = (unsigned short*)(ws + 3735808);
  unsigned short* hbuf    = (unsigned short*)(ws + 3768576); // 16 MB bf16 h

  zero_i32<<<NN / 256, 256, 0, stream>>>(counts, NN);
  hist_kernel<<<(NE + 255) / 256, 256, 0, stream>>>(dstv, counts);
  scan_kernel<<<1, 1024, 0, stream>>>(counts, row_ptr, cursor);
  scatter_kernel<<<(NE + 255) / 256, 256, 0, stream>>>(srcv, dstv, sel, cursor, csr);
  wprep_kernel<<<2 * KT1 + KT3, 256, 0, stream>>>(W1, W2, W3, Wf1, Wf2, Wf3);

  const size_t lds1 = (size_t)16 * KAGG1 * 4 + (size_t)16 * (KAGG1 + 8) * 2;  // 110848
  conv1_kernel<<<NN / 16, 128, lds1, stream>>>(x, row_ptr, csr, Wf1,
                                               b1, g1, be1, rm1, rv1, hbuf);

  const size_t lds2 = (size_t)16 * KAGG2 * 4 + (size_t)16 * (KAGG2 + 8) * 2;  // 123136
  conv2_kernel<<<NN / 16, 128, lds2, stream>>>(hbuf, x, row_ptr, csr, Wf2, Wf3,
                                               b2, g2, be2, rm2, rv2,
                                               b3, g3, be3, rm3, rv3,
                                               (float*)d_out);
}
